// Point_Enhance_7808250544222
// MI455X (gfx1250) — matrix operand order repaired and checked by reading, compiled, not run on hardware
//
#include <hip/hip_runtime.h>
#include <cstdint>

// Point-Enhance for MI455X (gfx1250, wave32).
// Big MLP on v_wmma_f32_16x16x32_f16 (f32 accum). Weights pre-swizzled to
// fragment order (every B operand = one contiguous 32B LDS load). B staging is
// double-buffered and copied with GLOBAL_LOAD_ASYNC_TO_LDS_B128 (ASYNCcnt),
// so global fetch fully overlaps WMMA issue with a single s_wait_asynccnt.

#define Nn    4
#define Cc    256
#define Hh    128
#define Ww    128
#define HW    (Hh * Ww)
#define Pp    256
#define KTOPK 1024
#define PH    64
#define CM    512   // C * MULT

typedef __attribute__((ext_vector_type(16))) _Float16 v16h;
typedef __attribute__((ext_vector_type(8)))  _Float16 v8h;
typedef __attribute__((ext_vector_type(8)))  float    v8f;

#define WMMA_F16(a, b, c) \
  __builtin_amdgcn_wmma_f32_16x16x32_f16(false, (a), false, (b), (short)0, (c), false, false)

// ---- async memory->LDS copy (CDNA5 ASYNCcnt path) ---------------------------
// GVS mode: lds_byte_addr (VGPR), 32-bit per-lane byte offset (VGPR), base (SGPR pair).
__device__ __forceinline__ void async_cp16(uint32_t lds_addr, const void* gbase, uint32_t goff) {
  asm volatile("global_load_async_to_lds_b128 %0, %1, %2"
               :: "v"(lds_addr), "v"(goff), "s"(gbase)
               : "memory");
}
__device__ __forceinline__ void async_wait0() {
  asm volatile("s_wait_asynccnt 0" ::: "memory");
}
// Low 32 bits of a generic pointer to LDS == LDS byte offset (ISA aperture rule).
__device__ __forceinline__ uint32_t lds_addr_of(const void* p) {
  return (uint32_t)(uintptr_t)p;
}

// ---- fragment helpers (ISA 7.12.2 layouts) ----------------------------------
__device__ __forceinline__ v16h ld_frag_a(const _Float16* tile, int ld, int lane) {
  int M = lane & 15, kb = (lane & 16) >> 1;   // kb: 0 or 8
  const _Float16* p = tile + (size_t)M * ld + kb;
  v8h lo = *(const v8h*)p;
  v8h hi = *(const v8h*)(p + 16);
  v16h f;
#pragma unroll
  for (int h = 0; h < 8; ++h) { f[h] = lo[h]; f[h + 8] = hi[h]; }
  return f;
}
__device__ __forceinline__ v16h ld_frag_b_swz(const _Float16* tile512, int lane) {
  return *(const v16h*)(tile512 + lane * 16);
}
__device__ __forceinline__ void st_frag_f16(_Float16* tile, int ld, v8f c, float bias,
                                            int lane, bool relu) {
  int Ncol = lane & 15, m0 = (lane & 16) >> 1;
#pragma unroll
  for (int r = 0; r < 8; ++r) {
    float v = c[r] + bias;
    if (relu) v = fmaxf(v, 0.f);
    tile[(size_t)(m0 + r) * ld + Ncol] = (_Float16)v;
  }
}
__device__ __forceinline__ void st_frag_f32(float* tile, int ld, v8f c, float bias, int lane) {
  int Ncol = lane & 15, m0 = (lane & 16) >> 1;
#pragma unroll
  for (int r = 0; r < 8; ++r) tile[(size_t)(m0 + r) * ld + Ncol] = c[r] + bias;
}

// B-operand swizzle: (k, n) of a KxN matrix -> fragment-ordered offset.
__device__ __forceinline__ size_t swz_b(int k, int n, int Ntot) {
  size_t tile = (size_t)((k >> 5) * (Ntot >> 4) + (n >> 4));
  return (tile << 9) + ((((k & 8) << 1) | (n & 15)) << 4) + ((k & 7) | ((k >> 1) & 8));
}

// ---- 1) stable top-k ranks (ties -> lower index, matching lax.top_k) --------
__global__ void __launch_bounds__(256) topk_rank_kernel(const float* __restrict__ edge,
                                                        int* __restrict__ rank2idx) {
  __shared__ float vals[HW];          // 64 KB
  int b = blockIdx.x >> 6;
  int chunk = blockIdx.x & 63;
  const float* e = edge + (size_t)b * HW;
  for (int t = threadIdx.x; t < HW; t += blockDim.x) vals[t] = e[t];
  __syncthreads();
  int idx = chunk * 256 + threadIdx.x;
  float v = vals[idx];
  int rank = 0;
  for (int j = 0; j < HW; ++j) {
    float u = vals[j];
    rank += (u > v) || (u == v && j < idx);
  }
  if (rank < KTOPK) rank2idx[(size_t)b * KTOPK + rank] = idx;
}

// ---- 2) select points, positions, deterministic "last dup wins" flag --------
__global__ void select_kernel(const int* __restrict__ rank2idx, const int* __restrict__ sel,
                              int* __restrict__ point_idx, int* __restrict__ winner,
                              float* __restrict__ pos) {
  int t = blockIdx.x * blockDim.x + threadIdx.x;
  if (t >= Nn * Pp) return;
  int b = t / Pp, p = t % Pp;
  int pi = rank2idx[(size_t)b * KTOPK + sel[p]];
  point_idx[t] = pi;
  pos[2 * t + 0] = (float)(pi % Hh) / (float)Ww;
  pos[2 * t + 1] = (float)(pi / Ww) / (float)Hh;
  int wn = 1;
  for (int p2 = p + 1; p2 < Pp; ++p2)
    if (rank2idx[(size_t)b * KTOPK + sel[p2]] == pi) { wn = 0; break; }
  winner[t] = wn;
}

// ---- 3) fp32 -> fp16 with B-fragment swizzle --------------------------------
__global__ void cvt_swz_kernel(const float* __restrict__ src, _Float16* __restrict__ dst,
                               int K, int Ntot) {
  int t = blockIdx.x * blockDim.x + threadIdx.x;
  if (t >= K * Ntot) return;
  int k = t / Ntot, n = t % Ntot;
  dst[swz_b(k, n, Ntot)] = (_Float16)src[t];
}

// ---- 4) gather x and q/k/v = x @ W (WMMA, async-staged swizzled W) ----------
__device__ __forceinline__ void qkv_mat(const v16h* af, const _Float16* __restrict__ Wswz,
                                        _Float16* __restrict__ dst0,
                                        _Float16* bst0, _Float16* bst1,
                                        int lane, int wave, int tid) {
  v8f acc[4] = {};
  uint32_t l0 = lds_addr_of(bst0), l1 = lds_addr_of(bst1);
#pragma unroll
  for (int u = 0; u < 8; ++u)
    async_cp16(l0 + (tid + u * 128) * 16, Wswz, (tid + u * 128) * 16);
  async_wait0();
  __syncthreads();
  for (int kc = 0; kc < 8; ++kc) {
    const _Float16* cur = (kc & 1) ? bst1 : bst0;
    if (kc + 1 < 8) {
      uint32_t ln = (kc & 1) ? l0 : l1;
      const char* gs = (const char*)Wswz + (size_t)(kc + 1) * 16384;
#pragma unroll
      for (int u = 0; u < 8; ++u)
        async_cp16(ln + (tid + u * 128) * 16, gs, (tid + u * 128) * 16);
    }
#pragma unroll
    for (int s = 0; s < 4; ++s)
      acc[s] = WMMA_F16(af[kc], ld_frag_b_swz(cur + (wave * 4 + s) * 512, lane), acc[s]);
    async_wait0();
    __syncthreads();
  }
  int Ncol = lane & 15, m0 = (lane & 16) >> 1;
#pragma unroll
  for (int s = 0; s < 4; ++s) {
    int ct = wave * 4 + s;
#pragma unroll
    for (int r = 0; r < 8; ++r)
      dst0[(size_t)(m0 + r) * Cc + ct * 16 + Ncol] = (_Float16)acc[s][r];
  }
  __syncthreads();
}

__global__ void __launch_bounds__(128) qkv_kernel(
    const float* __restrict__ feature, const int* __restrict__ point_idx,
    const _Float16* __restrict__ Wq, const _Float16* __restrict__ Wk,
    const _Float16* __restrict__ Wv,
    _Float16* __restrict__ qo, _Float16* __restrict__ ko, _Float16* __restrict__ vo) {
  __shared__ _Float16 xs[16 * Cc];        // 8 KB
  __shared__ _Float16 bst0[32 * Cc];      // 16 KB
  __shared__ _Float16 bst1[32 * Cc];      // 16 KB
  int b  = blockIdx.x / (Pp / 16);
  int p0 = (blockIdx.x % (Pp / 16)) * 16;
  for (int t = threadIdx.x; t < 16 * Cc; t += blockDim.x) {
    int r = t >> 8, c = t & 255;
    int pi = point_idx[b * Pp + p0 + r];
    xs[r * Cc + c] = (_Float16)feature[((size_t)b * Cc + c) * HW + pi];
  }
  __syncthreads();
  int tid = threadIdx.x, lane = tid & 31, wave = tid >> 5;
  v16h af[8];
#pragma unroll
  for (int kc = 0; kc < 8; ++kc) af[kc] = ld_frag_a(xs + kc * 32, Cc, lane);
  size_t row0 = ((size_t)b * Pp + p0) * Cc;
  qkv_mat(af, Wq, qo + row0, bst0, bst1, lane, wave, tid);
  qkv_mat(af, Wk, ko + row0, bst0, bst1, lane, wave, tid);
  qkv_mat(af, Wv, vo + row0, bst0, bst1, lane, wave, tid);
}

// ---- 5) fused rpe + MLP + online softmax + aggregation ----------------------
// One workgroup per (b, i); 8 waves; j in 4 blocks of 64. ~295 KB LDS.
__global__ void __launch_bounds__(256) attn_kernel(
    const int* __restrict__ point_idx, const float* __restrict__ pos,
    const _Float16* __restrict__ q16, const _Float16* __restrict__ k16,
    const _Float16* __restrict__ v16,
    const float* __restrict__ pw1, const float* __restrict__ pb1,
    const _Float16* __restrict__ pw2s, const float* __restrict__ pb2,
    const _Float16* __restrict__ aw1s, const float* __restrict__ ab1,
    const _Float16* __restrict__ aw2s, const float* __restrict__ ab2,
    float* __restrict__ aggr) {
  extern __shared__ char smem[];
  _Float16* posh = (_Float16*)smem;              // 256x64   16384 h
  _Float16* rpe  = posh + 256 * PH;              // 64x256   16384 h
  _Float16* tsh  = rpe  + 64 * Cc;               // 64x256   16384 h
  _Float16* hsh  = tsh  + 64 * Cc;               // 64x512   32768 h
  _Float16* bst0 = hsh  + (size_t)64 * CM;       // 16384 h (32 KB)
  _Float16* bst1 = bst0 + 16384;                 // 16384 h (32 KB)
  float*    sim  = (float*)(bst1 + 16384);       // 64x256 f32 (64 KB)
  _Float16* qsh  = (_Float16*)(sim + 64 * Cc);   // 256 h

  int b = blockIdx.x / Pp, i = blockIdx.x % Pp;
  int tid = threadIdx.x, lane = tid & 31, wave = tid >> 5;
  int rt = wave & 3;
  uint32_t l0 = lds_addr_of(bst0), l1 = lds_addr_of(bst1);

  qsh[tid] = q16[((size_t)b * Pp + i) * Cc + tid];
  float xi = pos[2 * (b * Pp + i)], yi = pos[2 * (b * Pp + i) + 1];
  {
    float xj = pos[2 * (b * Pp + tid)], yj = pos[2 * (b * Pp + tid) + 1];
    float dx = xi - xj, dy = yi - yj;
#pragma unroll 4
    for (int hh = 0; hh < PH; ++hh) {
      float u = fmaf(dx, pw1[hh], fmaf(dy, pw1[PH + hh], pb1[hh]));
      posh[tid * PH + hh] = (_Float16)fmaxf(u, 0.f);
    }
  }
  float m_r = -1e30f, l_r = 0.f, a_r = 0.f;   // online softmax state for d = tid
  __syncthreads();

  for (int jb = 0; jb < 4; ++jb) {
    int j0 = jb * 64;
    // -- rpe = posh @ pw2 + pb2 : async-stage whole swizzled pw2 (32 KB) --
#pragma unroll
    for (int u = 0; u < 8; ++u)
      async_cp16(l0 + (tid + u * 256) * 16, pw2s, (tid + u * 256) * 16);
    async_wait0();
    __syncthreads();
    {
      v16h a0 = ld_frag_a(posh + (size_t)(j0 + rt * 16) * PH + 0,  PH, lane);
      v16h a1 = ld_frag_a(posh + (size_t)(j0 + rt * 16) * PH + 32, PH, lane);
      int cb = (wave >> 2) * 8;
      for (int s = 0; s < 8; ++s) {
        int ct = cb + s;
        v8f acc = {};
        acc = WMMA_F16(a0, ld_frag_b_swz(bst0 + ct * 512, lane), acc);          // kblk 0
        acc = WMMA_F16(a1, ld_frag_b_swz(bst0 + 8192 + ct * 512, lane), acc);   // kblk 1
        st_frag_f16(rpe + (size_t)rt * 16 * Cc + ct * 16, Cc, acc,
                    pb2[ct * 16 + (lane & 15)], lane, false);
      }
    }
    __syncthreads();
    // -- t = q_i - k_j + rpe (packed f16 vector math) --
    {
      int jj = tid >> 2, c0 = (tid & 3) * 64;
      const _Float16* krow = k16 + ((size_t)b * Pp + j0 + jj) * Cc;
#pragma unroll
      for (int u = 0; u < 8; ++u) {
        int d0 = c0 + u * 8;
        v8h qh = *(const v8h*)(qsh + d0);
        v8h kv = *(const v8h*)(krow + d0);
        v8h rp = *(const v8h*)(rpe + (size_t)jj * Cc + d0);
        *(v8h*)(tsh + (size_t)jj * Cc + d0) = qh - kv + rp;
      }
    }
    // -- GEMM1: h = relu(t @ aw1 + ab1), 64x512, async double-buffered B --
    {
      int cb = (wave >> 2) * 16;
      v8f acc[16] = {};
#pragma unroll
      for (int u = 0; u < 8; ++u)
        async_cp16(l0 + (tid + u * 256) * 16, aw1s, (tid + u * 256) * 16);
      async_wait0();
      __syncthreads();
      for (int kc = 0; kc < 8; ++kc) {
        const _Float16* cur = (kc & 1) ? bst1 : bst0;
        if (kc + 1 < 8) {
          uint32_t ln = (kc & 1) ? l0 : l1;
          const char* gs = (const char*)aw1s + (size_t)(kc + 1) * 32768;
#pragma unroll
          for (int u = 0; u < 8; ++u)
            async_cp16(ln + (tid + u * 256) * 16, gs, (tid + u * 256) * 16);
        }
        v16h af = ld_frag_a(tsh + (size_t)rt * 16 * Cc + kc * 32, Cc, lane);
#pragma unroll
        for (int s = 0; s < 16; ++s)
          acc[s] = WMMA_F16(af, ld_frag_b_swz(cur + (cb + s) * 512, lane), acc[s]);
        async_wait0();
        __syncthreads();
      }
#pragma unroll
      for (int s = 0; s < 16; ++s) {
        int ct = cb + s;
        st_frag_f16(hsh + (size_t)rt * 16 * CM + ct * 16, CM, acc[s],
                    ab1[ct * 16 + (lane & 15)], lane, true);
      }
    }
    __syncthreads();
    // -- GEMM2: sim = h @ aw2 + ab2, 64x256 f32, async double-buffered B --
    {
      int cb = (wave >> 2) * 8;
      v8f acc[8] = {};
#pragma unroll
      for (int u = 0; u < 4; ++u)
        async_cp16(l0 + (tid + u * 256) * 16, aw2s, (tid + u * 256) * 16);
      async_wait0();
      __syncthreads();
      for (int kc = 0; kc < 16; ++kc) {
        const _Float16* cur = (kc & 1) ? bst1 : bst0;
        if (kc + 1 < 16) {
          uint32_t ln = (kc & 1) ? l0 : l1;
          const char* gs = (const char*)aw2s + (size_t)(kc + 1) * 16384;
#pragma unroll
          for (int u = 0; u < 4; ++u)
            async_cp16(ln + (tid + u * 256) * 16, gs, (tid + u * 256) * 16);
        }
        v16h af = ld_frag_a(hsh + (size_t)rt * 16 * CM + kc * 32, CM, lane);
#pragma unroll
        for (int s = 0; s < 8; ++s)
          acc[s] = WMMA_F16(af, ld_frag_b_swz(cur + (cb + s) * 512, lane), acc[s]);
        async_wait0();
        __syncthreads();
      }
#pragma unroll
      for (int s = 0; s < 8; ++s) {
        int ct = cb + s;
        st_frag_f32(sim + (size_t)rt * 16 * Cc + ct * 16, Cc, acc[s],
                    ab2[ct * 16 + (lane & 15)], lane);
      }
    }
    __syncthreads();
    // -- online softmax over this j-block (thread owns channel d = tid) --
    {
      float mb = m_r;
      for (int jj = 0; jj < 64; ++jj) mb = fmaxf(mb, sim[(size_t)jj * Cc + tid]);
      float sc = __expf(m_r - mb);
      l_r *= sc; a_r *= sc;
      for (int jj = 0; jj < 64; ++jj) {
        float e  = __expf(sim[(size_t)jj * Cc + tid] - mb);
        float vv = (float)v16[((size_t)b * Pp + j0 + jj) * Cc + tid]
                 + (float)rpe[(size_t)jj * Cc + tid];
        l_r += e;
        a_r = fmaf(e, vv, a_r);
      }
      m_r = mb;
    }
    __syncthreads();
  }
  aggr[((size_t)b * Cc + tid) * Pp + i] = a_r / l_r;   // (n, C, P)
}

// ---- 6) scatter winners into output copy of feature -------------------------
__global__ void __launch_bounds__(256) scatter_kernel(
    const float* __restrict__ aggr, const int* __restrict__ point_idx,
    const int* __restrict__ winner, float* __restrict__ out) {
  int bp = blockIdx.x;
  if (!winner[bp]) return;
  int b = bp / Pp, p = bp % Pp, c = threadIdx.x;
  out[((size_t)b * Cc + c) * HW + point_idx[bp]] =
      aggr[((size_t)b * Cc + c) * Pp + p];
}

extern "C" void kernel_launch(void* const* d_in, const int* in_sizes, int n_in,
                              void* d_out, int out_size, void* d_ws, size_t ws_size,
                              hipStream_t stream) {
  (void)in_sizes; (void)n_in; (void)out_size; (void)ws_size;
  const float* edge    = (const float*)d_in[0];
  const float* feature = (const float*)d_in[1];
  const int*   sel     = (const int*)d_in[2];
  const float* Wq  = (const float*)d_in[3];
  const float* Wk  = (const float*)d_in[4];
  const float* Wv  = (const float*)d_in[5];
  const float* pw1 = (const float*)d_in[6];
  const float* pb1 = (const float*)d_in[7];
  const float* pw2 = (const float*)d_in[8];
  const float* pb2 = (const float*)d_in[9];
  const float* aw1 = (const float*)d_in[10];
  const float* ab1 = (const float*)d_in[11];
  const float* aw2 = (const float*)d_in[12];
  const float* ab2 = (const float*)d_in[13];
  float* out = (float*)d_out;

  // workspace carve-out (~3.5 MB)
  char* w = (char*)d_ws;
  size_t off = 0;
  auto take = [&](size_t bytes) -> char* {
    char* p = w + off;
    off += (bytes + 255) & ~(size_t)255;
    return p;
  };
  int*      rank2idx  = (int*)take((size_t)Nn * KTOPK * 4);
  int*      point_idx = (int*)take((size_t)Nn * Pp * 4);
  int*      winner    = (int*)take((size_t)Nn * Pp * 4);
  float*    posw      = (float*)take((size_t)Nn * Pp * 2 * 4);
  _Float16* q16  = (_Float16*)take((size_t)Nn * Pp * Cc * 2);
  _Float16* k16  = (_Float16*)take((size_t)Nn * Pp * Cc * 2);
  _Float16* v16  = (_Float16*)take((size_t)Nn * Pp * Cc * 2);
  _Float16* Wq16 = (_Float16*)take((size_t)Cc * Cc * 2);
  _Float16* Wk16 = (_Float16*)take((size_t)Cc * Cc * 2);
  _Float16* Wv16 = (_Float16*)take((size_t)Cc * Cc * 2);
  _Float16* pw2f = (_Float16*)take((size_t)PH * Cc * 2);
  _Float16* aw1f = (_Float16*)take((size_t)Cc * CM * 2);
  _Float16* aw2f = (_Float16*)take((size_t)CM * Cc * 2);
  float*    aggr = (float*)take((size_t)Nn * Cc * Pp * 4);

  hipMemcpyAsync(out, feature, (size_t)Nn * Cc * HW * sizeof(float),
                 hipMemcpyDeviceToDevice, stream);

  topk_rank_kernel<<<Nn * 64, 256, 0, stream>>>(edge, rank2idx);
  select_kernel<<<(Nn * Pp + 255) / 256, 256, 0, stream>>>(rank2idx, sel, point_idx, winner, posw);

  cvt_swz_kernel<<<(Cc * Cc + 255) / 256, 256, 0, stream>>>(Wq, Wq16, Cc, Cc);
  cvt_swz_kernel<<<(Cc * Cc + 255) / 256, 256, 0, stream>>>(Wk, Wk16, Cc, Cc);
  cvt_swz_kernel<<<(Cc * Cc + 255) / 256, 256, 0, stream>>>(Wv, Wv16, Cc, Cc);
  cvt_swz_kernel<<<(PH * Cc + 255) / 256, 256, 0, stream>>>(pw2, pw2f, PH, Cc);
  cvt_swz_kernel<<<(Cc * CM + 255) / 256, 256, 0, stream>>>(aw1, aw1f, Cc, CM);
  cvt_swz_kernel<<<(CM * Cc + 255) / 256, 256, 0, stream>>>(aw2, aw2f, CM, Cc);

  qkv_kernel<<<Nn * (Pp / 16), 128, 0, stream>>>(feature, point_idx, Wq16, Wk16, Wv16,
                                                 q16, k16, v16);

  size_t lds = (size_t)(256 * PH + 64 * Cc + 64 * Cc + 64 * CM + 2 * 16384 + 256) * 2
             + (size_t)(64 * Cc) * 4;   // 295,424 B < 320 KB
  attn_kernel<<<Nn * Pp, 256, lds, stream>>>(point_idx, posw, q16, k16, v16,
                                             pw1, pb1, pw2f, pb2,
                                             aw1f, ab1, aw2f, ab2, aggr);

  scatter_kernel<<<Nn * Pp, Cc, 0, stream>>>(aggr, point_idx, winner, out);
}
